// LovaszHingeLoss_59863254172452
// MI455X (gfx1250) — matrix operand order repaired and checked by reading, not compiled, not run on hardware
//
#include <hip/hip_runtime.h>
#include <hip/hip_bf16.h>

// ---------------------------------------------------------------------------
// Lovasz hinge loss, MI455X (gfx1250).
// Pipeline:
//   k_build    : pred/target -> (orderable key u32, label u8)
//   4x radix-8 : k_count -> k_scan_hist -> k_scatter (stable, LDS rank matrix,
//                tiles staged with global_load_async_to_lds_b128)
//   k_bsum/k_bscan : per-block label sums + per-image exclusive block offsets
//   k_loss     : WMMA (v_wmma_f32_16x16x32_f16) 256-way block scan of labels,
//                telescoped Jaccard-gradient dot product, per-block partials
//   k_final    : deterministic tree reduction -> d_out[0] = sum / 64
// ---------------------------------------------------------------------------

typedef __attribute__((ext_vector_type(16))) _Float16 v16h;
typedef __attribute__((ext_vector_type(8)))  float    v8f;

#define NIMG   64
#define NPIX   262144          // 512*512
#define TOTAL  (NIMG * NPIX)   // 2^24
#define TILE   4096            // elements per block
#define BPI    64              // blocks per image (NPIX / TILE)
#define THR    256
#define IPT    16              // items per thread

// ---- key transform: ascending sort of key <=> descending sort of error ----
__device__ __forceinline__ unsigned key_from_err(float e) {
  unsigned u = __float_as_uint(e);
  unsigned asc = u ^ ((unsigned)((int)u >> 31) | 0x80000000u); // ascending-orderable
  return ~asc;                                                 // invert: descending error
}
__device__ __forceinline__ float err_from_key(unsigned key) {
  unsigned a = ~key;
  unsigned u = (a & 0x80000000u) ? (a ^ 0x80000000u) : ~a;
  return __uint_as_float(u);
}

// ---------------------------------------------------------------------------
__global__ __launch_bounds__(THR) void k_build(const float* __restrict__ pred,
                                               const float* __restrict__ tgt,
                                               unsigned* __restrict__ keys,
                                               unsigned char* __restrict__ vals) {
  int i = (blockIdx.x * THR + threadIdx.x) * 4;
  float4 p = *(const float4*)(pred + i);
  float4 t = *(const float4*)(tgt + i);
  unsigned k0 = key_from_err(fmaf(-p.x, 2.0f * t.x - 1.0f, 1.0f));
  unsigned k1 = key_from_err(fmaf(-p.y, 2.0f * t.y - 1.0f, 1.0f));
  unsigned k2 = key_from_err(fmaf(-p.z, 2.0f * t.z - 1.0f, 1.0f));
  unsigned k3 = key_from_err(fmaf(-p.w, 2.0f * t.w - 1.0f, 1.0f));
  uint4 kk = make_uint4(k0, k1, k2, k3);
  *(uint4*)(keys + i) = kk;
  unsigned vv = (unsigned)(t.x != 0.0f) | ((unsigned)(t.y != 0.0f) << 8) |
                ((unsigned)(t.z != 0.0f) << 16) | ((unsigned)(t.w != 0.0f) << 24);
  *(unsigned*)(vals + i) = vv;
}

// ---------------------------------------------------------------------------
__global__ __launch_bounds__(THR) void k_count(const unsigned* __restrict__ keys,
                                               unsigned* __restrict__ hist, int shift) {
  __shared__ unsigned lh[256];
  int t = threadIdx.x;
  lh[t] = 0;
  __syncthreads();
  int img = blockIdx.x >> 6, blk = blockIdx.x & 63;
  const uint4* b4 = (const uint4*)(keys + ((size_t)img * NPIX + (size_t)blk * TILE)) + t * 4;
#pragma unroll
  for (int q = 0; q < 4; ++q) {
    uint4 k = b4[q];
    atomicAdd(&lh[(k.x >> shift) & 255u], 1u);
    atomicAdd(&lh[(k.y >> shift) & 255u], 1u);
    atomicAdd(&lh[(k.z >> shift) & 255u], 1u);
    atomicAdd(&lh[(k.w >> shift) & 255u], 1u);
  }
  __syncthreads();
  hist[((img * 256) + t) * 64 + blk] = lh[t];  // digit-major per image
}

// per-image exclusive scan over (digit, block), in place -------------------
__global__ __launch_bounds__(THR) void k_scan_hist(unsigned* __restrict__ hist) {
  __shared__ unsigned sh[256];
  int img = blockIdx.x, t = threadIdx.x;
  unsigned* hb = hist + (size_t)img * 16384;
  unsigned s = 0;
  for (int k = 0; k < 64; ++k) s += hb[t * 64 + k];
  sh[t] = s;
  __syncthreads();
  if (t == 0) {
    unsigned run = 0;
    for (int q = 0; q < 256; ++q) { unsigned v = sh[q]; sh[q] = run; run += v; }
  }
  __syncthreads();
  unsigned run = sh[t];
  for (int k = 0; k < 64; ++k) { unsigned v = hb[t * 64 + k]; hb[t * 64 + k] = run; run += v; }
}

// stable scatter: async-staged tile + 256x256 u16 rank matrix in LDS --------
__global__ __launch_bounds__(THR) void k_scatter(const unsigned* __restrict__ keysIn,
                                                 const unsigned char* __restrict__ valsIn,
                                                 unsigned* __restrict__ keysOut,
                                                 unsigned char* __restrict__ valsOut,
                                                 const unsigned* __restrict__ hist, int shift) {
  __shared__ __align__(16) unsigned       sh_keys[TILE];        // 16 KB
  __shared__ __align__(16) unsigned char  sh_vals[TILE];        // 4 KB
  __shared__ unsigned short               cnt[256 * 256];       // 128 KB rank matrix
  __shared__ unsigned                     sh_base[256];         // 1 KB
  int t = threadIdx.x;
  int img = blockIdx.x >> 6, blk = blockIdx.x & 63;
  size_t tileStart = (size_t)img * NPIX + (size_t)blk * TILE;

  // --- CDNA5 async tensor-path staging: global -> LDS, bypassing VGPRs ---
  {
    unsigned long long gk = (unsigned long long)(const void*)(keysIn + tileStart);
    unsigned long long gv = (unsigned long long)(const void*)(valsIn + tileStart);
    unsigned lk = (unsigned)(unsigned long long)(const void*)&sh_keys[0];
    unsigned lv = (unsigned)(unsigned long long)(const void*)&sh_vals[0];
#pragma unroll
    for (int q = 0; q < 4; ++q) {
      unsigned off = (unsigned)(q * THR + t) * 16u;
      unsigned long long ga = gk + off;
      unsigned la = lk + off;
      asm volatile("global_load_async_to_lds_b128 %0, %1, off" :: "v"(la), "v"(ga) : "memory");
    }
    {
      unsigned off = (unsigned)t * 16u;
      unsigned long long ga = gv + off;
      unsigned la = lv + off;
      asm volatile("global_load_async_to_lds_b128 %0, %1, off" :: "v"(la), "v"(ga) : "memory");
    }
    asm volatile("s_wait_asynccnt 0" ::: "memory");
  }
  // zero rank matrix (each thread zeroes its own 256-entry row, as dwords)
  unsigned* cz = (unsigned*)cnt;
#pragma unroll 8
  for (int q = 0; q < 128; ++q) cz[t * 128 + q] = 0;
  // preload scatter bases for all 256 digits of this block
  sh_base[t] = (unsigned)(img * NPIX) + hist[((img * 256) + t) * 64 + blk];
  __syncthreads();

  // phase 1: per-thread digit counts (thread t owns contiguous items -> stable)
  unsigned char digs[IPT];
#pragma unroll
  for (int j = 0; j < IPT; ++j) {
    unsigned k = sh_keys[t * IPT + j];
    unsigned d = (k >> shift) & 255u;
    digs[j] = (unsigned char)d;
    cnt[t * 256 + d] = (unsigned short)(cnt[t * 256 + d] + 1);
  }
  __syncthreads();
  // phase 2: thread d performs exclusive scan down column d (over 256 threads)
  {
    unsigned run = 0;
    for (int q = 0; q < 256; ++q) {
      unsigned v = cnt[q * 256 + t];
      cnt[q * 256 + t] = (unsigned short)run;
      run += v;
    }
  }
  __syncthreads();
  // phase 3: stable scatter
#pragma unroll
  for (int j = 0; j < IPT; ++j) {
    unsigned d = (unsigned)digs[j];
    unsigned r = cnt[t * 256 + d];
    cnt[t * 256 + d] = (unsigned short)(r + 1);
    unsigned pos = sh_base[d] + r;
    keysOut[pos] = sh_keys[t * IPT + j];
    valsOut[pos] = sh_vals[t * IPT + j];
  }
}

// ---------------------------------------------------------------------------
__global__ __launch_bounds__(THR) void k_bsum(const unsigned char* __restrict__ vals,
                                              unsigned* __restrict__ bsum) {
  __shared__ unsigned sh[256];
  int t = threadIdx.x;
  int img = blockIdx.x >> 6, blk = blockIdx.x & 63;
  uint4 v = *((const uint4*)(vals + ((size_t)img * NPIX + (size_t)blk * TILE)) + t);
  unsigned s = 0;
#pragma unroll
  for (int q = 0; q < 4; ++q) {
    unsigned w = (&v.x)[q];
    s += (w & 0xFFu) + ((w >> 8) & 0xFFu) + ((w >> 16) & 0xFFu) + (w >> 24);
  }
  sh[t] = s;
  __syncthreads();
  for (int st = 128; st > 0; st >>= 1) {
    if (t < st) sh[t] += sh[t + st];
    __syncthreads();
  }
  if (t == 0) bsum[blockIdx.x] = sh[0];
}

__global__ __launch_bounds__(64) void k_bscan(const unsigned* __restrict__ bsum,
                                              float* __restrict__ boff,
                                              float* __restrict__ Ptot) {
  __shared__ unsigned sh[64];
  int img = blockIdx.x, t = threadIdx.x;
  sh[t] = bsum[img * 64 + t];
  __syncthreads();
  if (t == 0) {
    unsigned run = 0;
    for (int q = 0; q < 64; ++q) { unsigned v = sh[q]; sh[q] = run; run += v; }
    Ptot[img] = (float)run;
  }
  __syncthreads();
  boff[img * 64 + t] = (float)sh[t];
}

// ---------------------------------------------------------------------------
// loss kernel: WMMA-based 256-way scan of per-thread label sums.
__global__ __launch_bounds__(THR) void k_loss(const unsigned* __restrict__ keys,
                                              const unsigned char* __restrict__ vals,
                                              const float* __restrict__ boff,
                                              const float* __restrict__ Ptot,
                                              float* __restrict__ partials) {
  __shared__ __align__(16) float sh_tsum[256];
  __shared__ __align__(16) float sh_ST[256];   // S transposed: sh_ST[c*16 + k] = S[k, c]
  __shared__ float sh_off[16];
  __shared__ float sh_excl[256];
  __shared__ float sh_red[256];
  int t = threadIdx.x;
  int img = blockIdx.x >> 6, blk = blockIdx.x & 63;
  int elemBase = blk * TILE + t * IPT;          // position within image
  size_t gBase = (size_t)img * NPIX + (size_t)elemBase;

  // labels + within-thread inclusive scan
  uint4 vv = *((const uint4*)(vals + gBase));
  float incl[IPT];
  float run = 0.0f;
#pragma unroll
  for (int j = 0; j < IPT; ++j) {
    unsigned w = (&vv.x)[j >> 2];
    float g = (float)((w >> ((j & 3) * 8)) & 0xFFu);
    run += g;
    incl[j] = run;
  }
  sh_tsum[t] = run;
  __syncthreads();

  // ---- WMMA scan of 256 thread sums (each wave redundantly, EXEC all-1) ----
  {
    int lane = threadIdx.x & 31;
    int m = lane & 15;                 // row (A) / col (B,D)
    int kbase = (lane < 16) ? 0 : 8;   // A operand K-half owned by this lane
    int rbase = kbase;                 // D rows owned by this lane
    float inb = (lane < 16) ? 1.0f : 0.0f;  // value-mask (no exec divergence)
    v16h a, bU;
#pragma unroll
    for (int j = 0; j < 8; ++j) {
      a[j] = (_Float16)sh_tsum[m * 16 + kbase + j]; // X[m, k], k = kbase+j
      a[j + 8] = (_Float16)0.0f;                    // K=16..31 zero
    }
#pragma unroll
    for (int k = 0; k < 16; ++k)                    // U[kbase+k, m] = (kbase+k <= m) for k<8, K=16..31 zero
      bU[k] = (_Float16)((k < 8 && (kbase + k) <= m) ? 1.0f : 0.0f);
    v8f c0 = {};
    // S = X * U : inclusive row scans
    v8f S = __builtin_amdgcn_wmma_f32_16x16x32_f16(false, a, false, bU,
                                                   (short)0, c0, false, false);
    // store S transposed: lane writes 8 contiguous floats -> ds_store_b128 x2
    {
      float4 w0 = make_float4(S[0], S[1], S[2], S[3]);
      float4 w1 = make_float4(S[4], S[5], S[6], S[7]);
      float4* sp = (float4*)&sh_ST[m * 16 + rbase];
      sp[0] = w0;
      sp[1] = w1;
    }
    asm volatile("s_wait_dscnt 0" ::: "memory");

    v16h a2, b2;
#pragma unroll
    for (int j = 0; j < 8; ++j) {                   // Lstrict[m, k] = (k < m)
      a2[j] = (_Float16)(((kbase + j) < m) ? 1.0f : 0.0f);
      a2[j + 8] = (_Float16)0.0f;
    }
    // B2[k, c=m] = S[k, m] : contiguous 64B read, value-masked (branch-free)
    {
      const float4* sp = (const float4*)&sh_ST[m * 16];
      float4 s0 = sp[0], s1 = sp[1], s2 = sp[2], s3 = sp[3];
      float sv[16] = { s0.x, s0.y, s0.z, s0.w, s1.x, s1.y, s1.z, s1.w,
                       s2.x, s2.y, s2.z, s2.w, s3.x, s3.y, s3.z, s3.w };
#pragma unroll
      for (int k = 0; k < 16; ++k) b2[k] = (_Float16)((k < 8) ? sv[kbase + k] : 0.0f);
    }
    // T = Lstrict * S : T[r,15] = sum of row totals of rows < r
    v8f T = __builtin_amdgcn_wmma_f32_16x16x32_f16(false, a2, false, b2,
                                                   (short)0, c0, false, false);
    if (m == 15) {
#pragma unroll
      for (int j = 0; j < 8; ++j) sh_off[rbase + j] = T[j];
    }
    asm volatile("s_wait_dscnt 0" ::: "memory");
#pragma unroll
    for (int j = 0; j < 8; ++j) {
      int r = rbase + j;
      int tp = r * 16 + m;                          // flattened thread index
      sh_excl[tp] = S[j] + sh_off[r] - sh_tsum[tp]; // exclusive prefix
    }
  }
  __syncthreads();

  float excl = sh_excl[t];
  float bo = boff[img * 64 + blk];
  float P = Ptot[img];

  // keys (16 own + 1 neighbor) -> relu(errors_sorted)
  unsigned karr[IPT];
  {
    const uint4* kp = (const uint4*)(keys + gBase);
#pragma unroll
    for (int q = 0; q < 4; ++q) {
      uint4 kk = kp[q];
      karr[q * 4 + 0] = kk.x; karr[q * 4 + 1] = kk.y;
      karr[q * 4 + 2] = kk.z; karr[q * 4 + 3] = kk.w;
    }
  }
  float Rnext = 0.0f;
  if (elemBase + IPT < NPIX) Rnext = fmaxf(err_from_key(keys[gBase + IPT]), 0.0f);

  float R[IPT + 1];
#pragma unroll
  for (int j = 0; j < IPT; ++j) R[j] = fmaxf(err_from_key(karr[j]), 0.0f);
  R[IPT] = Rnext;

  // loss = sum_i J_i * (R_i - R_{i+1}),  J_i = 1 - (P - p_i) / (P + (i+1) - p_i)
  float acc = 0.0f;
#pragma unroll
  for (int j = 0; j < IPT; ++j) {
    float p = bo + excl + incl[j];
    float rank1 = (float)(elemBase + j + 1);
    float uni = P + rank1 - p;
    float J = 1.0f - (P - p) / uni;
    acc += J * (R[j] - R[j + 1]);
  }
  sh_red[t] = acc;
  __syncthreads();
  for (int st = 128; st > 0; st >>= 1) {
    if (t < st) sh_red[t] += sh_red[t + st];
    __syncthreads();
  }
  if (t == 0) partials[blockIdx.x] = sh_red[0];
}

// ---------------------------------------------------------------------------
__global__ __launch_bounds__(THR) void k_final(const float* __restrict__ partials,
                                               float* __restrict__ out) {
  __shared__ float sh[256];
  int t = threadIdx.x;
  float s = 0.0f;
#pragma unroll
  for (int k = 0; k < 16; ++k) s += partials[k * 256 + t];
  sh[t] = s;
  __syncthreads();
  for (int st = 128; st > 0; st >>= 1) {
    if (t < st) sh[t] += sh[t + st];
    __syncthreads();
  }
  if (t == 0) out[0] = sh[0] * (1.0f / (float)NIMG);
}

// ---------------------------------------------------------------------------
extern "C" void kernel_launch(void* const* d_in, const int* in_sizes, int n_in,
                              void* d_out, int out_size, void* d_ws, size_t ws_size,
                              hipStream_t stream) {
  (void)in_sizes; (void)n_in; (void)out_size;
  const float* pred = (const float*)d_in[0];
  const float* tgt  = (const float*)d_in[1];
  float* out = (float*)d_out;

  // workspace carve-up (~164.1 MiB)
  unsigned char* ws = (unsigned char*)d_ws;
  unsigned*      keysA = (unsigned*)ws;                               // 64 MiB
  unsigned*      keysB = keysA + TOTAL;                               // 64 MiB
  unsigned char* valsA = (unsigned char*)(keysB + TOTAL);             // 16 MiB
  unsigned char* valsB = valsA + TOTAL;                               // 16 MiB
  unsigned*      hist  = (unsigned*)(valsB + TOTAL);                  // 4 MiB
  unsigned*      bsum  = hist + (size_t)NIMG * 256 * 64;
  float*         boff  = (float*)(bsum + NIMG * BPI);
  float*         Pt    = boff + NIMG * BPI;
  float*         parts = Pt + NIMG;
  if (ws_size < (size_t)(((unsigned char*)(parts + NIMG * BPI)) - ws)) return;

  const int nblk = NIMG * BPI;  // 4096

  k_build<<<TOTAL / (THR * 4), THR, 0, stream>>>(pred, tgt, keysA, valsA);

  unsigned* ki = keysA; unsigned char* vi = valsA;
  unsigned* ko = keysB; unsigned char* vo = valsB;
  for (int pass = 0; pass < 4; ++pass) {
    int shift = pass * 8;
    k_count<<<nblk, THR, 0, stream>>>(ki, hist, shift);
    k_scan_hist<<<NIMG, THR, 0, stream>>>(hist);
    k_scatter<<<nblk, THR, 0, stream>>>(ki, vi, ko, vo, hist, shift);
    unsigned* tk = ki; ki = ko; ko = tk;
    unsigned char* tv = vi; vi = vo; vo = tv;
  }
  // after 4 passes, sorted data is back in keysA/valsA (== ki/vi)

  k_bsum<<<nblk, THR, 0, stream>>>(vi, bsum);
  k_bscan<<<NIMG, 64, 0, stream>>>(bsum, boff, Pt);
  k_loss<<<nblk, THR, 0, stream>>>(ki, vi, boff, Pt, parts);
  k_final<<<1, THR, 0, stream>>>(parts, out);
}
